// LoraLinear_88931592831072
// MI455X (gfx1250) — matrix operand order repaired and checked by reading, not compiled, not run on hardware
//
#include <hip/hip_runtime.h>
#include <hip/hip_bf16.h>

typedef __bf16 bf16_t;
typedef __bf16 v16bf __attribute__((ext_vector_type(16)));
typedef __bf16 v8bf  __attribute__((ext_vector_type(8)));
typedef __bf16 v4bf  __attribute__((ext_vector_type(4)));
typedef float  v8f   __attribute__((ext_vector_type(8)));

#define M_TOT   16384   // B*S = 4*4096
#define K_TOT   1024    // DIN
#define N_TOT   1024    // DOUT
#define T_CNT   4
#define R_RANK  16

#define TILE_M  128
#define TILE_N  128
#define TILE_K  64
#define LDS_STRIDE 72   // 64 + 8 pad (bf16) -> 144B rows, de-aliases LDS banks

static __device__ __forceinline__ v16bf cat8(v8bf lo, v8bf hi) {
    return __builtin_shufflevector(lo, hi, 0,1,2,3,4,5,6,7,8,9,10,11,12,13,14,15);
}

// CDNA5 async global->LDS copy (ASYNCcnt-tracked, no VGPR staging).
// LDS byte address = low 32 bits of the flat pointer (LDS aperture occupies the
// high 32 bits per ISA 10.2).
static __device__ __forceinline__ void async_ldg_b128(bf16_t* lds_dst,
                                                      const bf16_t* gsrc) {
    unsigned lds_addr = (unsigned)(unsigned long long)lds_dst;
    asm volatile("global_load_async_to_lds_b128 %0, %1, off"
                 :: "v"(lds_addr), "v"(gsrc)
                 : "memory");
}
static __device__ __forceinline__ void wait_asynccnt_8() {
    asm volatile("s_wait_asynccnt 0x8" ::: "memory");
}
static __device__ __forceinline__ void wait_asynccnt_0() {
    asm volatile("s_wait_asynccnt 0x0" ::: "memory");
}

// ---------------------------------------------------------------------------
// Kernel 0a: convert x (f32) -> bf16, 4 elements/thread
// ---------------------------------------------------------------------------
__global__ __launch_bounds__(256) void cvt_x_bf16(const float* __restrict__ x,
                                                  bf16_t* __restrict__ xb) {
    size_t i = (size_t)blockIdx.x * 256 + threadIdx.x;   // index of float4 group
    float4 v = ((const float4*)x)[i];
    v4bf o;
    o[0] = (bf16_t)v.x; o[1] = (bf16_t)v.y; o[2] = (bf16_t)v.z; o[3] = (bf16_t)v.w;
    ((v4bf*)xb)[i] = o;
}

// ---------------------------------------------------------------------------
// Kernel 0b: Weff[t][n][k] = W[n][k] + sum_r B[ti][n][r]*A[ti][r][k]  (bf16 out)
// ---------------------------------------------------------------------------
__global__ __launch_bounds__(256) void build_weff(const float* __restrict__ W,
                                                  const float* __restrict__ A,
                                                  const float* __restrict__ Bm,
                                                  const int*   __restrict__ tuner,
                                                  bf16_t* __restrict__ weff) {
    size_t i = (size_t)blockIdx.x * 256 + threadIdx.x;   // over T*N*K = 4M
    int k = (int)(i & (K_TOT - 1));
    int n = (int)((i >> 10) & (N_TOT - 1));
    int t = (int)(i >> 20);
    int ti = tuner[t];
    float acc = W[(size_t)n * K_TOT + k];
    const float* Brow = Bm + ((size_t)ti * N_TOT + n) * R_RANK;
    const float* Acol = A + (size_t)ti * R_RANK * K_TOT + k;
#pragma unroll
    for (int r = 0; r < R_RANK; ++r)
        acc += Brow[r] * Acol[(size_t)r * K_TOT];
    weff[i] = (bf16_t)acc;
}

// ---------------------------------------------------------------------------
// Kernel 1: batched GEMM  out[t] = Xb @ Weff_t^T + bias
//   block = 256 threads (8 wave32), block tile 128x128, K-step 64
//   wave tile 32x64 = 2x4 v_wmma_f32_16x16x32_bf16 fragments (8 wmma/K-step,
//   16 wmma per staged tile). Double-buffered LDS, async global->LDS staging.
// ---------------------------------------------------------------------------
__global__ __launch_bounds__(256) void lora_gemm(const bf16_t* __restrict__ Xb,
                                                 const bf16_t* __restrict__ Weff,
                                                 const float*  __restrict__ bias,
                                                 float*        __restrict__ out) {
    __shared__ bf16_t lx[2 * TILE_M * LDS_STRIDE];   // 2 x 128 x (64+pad) bf16
    __shared__ bf16_t lw[2 * TILE_N * LDS_STRIDE];   // 2 x 128 x (64+pad) bf16
    const unsigned bufX = TILE_M * LDS_STRIDE;       // elements per buffer
    const unsigned bufW = TILE_N * LDS_STRIDE;

    const int m0 = blockIdx.x * TILE_M;
    const int n0 = blockIdx.y * TILE_N;
    const int t  = blockIdx.z;
    const bf16_t* Wt = Weff + (size_t)t * N_TOT * K_TOT;

    const int tid  = threadIdx.x;
    const int lane = tid & 31;
    const int wave = tid >> 5;
    const int wm   = wave >> 1;   // 0..3 (M direction, 32 rows each)
    const int wn   = wave & 1;    // 0..1 (N direction, 64 cols each)
    const int hl   = lane >> 4;   // half-wave selector
    const int l15  = lane & 15;

    // staging: tile row = 64 k elems = 128B = 8 x 16B chunks; 256 threads move
    // 4 chunks each for X and 4 for W  (8 async b128 per thread per tile)
    const int srow = tid >> 3;            // 0..31
    const int scol = (tid & 7) * 8;       // 0..56 (bf16 elems)
    const bf16_t* gx[4]; const bf16_t* gw[4];
    bf16_t* lxp[4]; bf16_t* lwp[4];
#pragma unroll
    for (int p = 0; p < 4; ++p) {
        int row = srow + 32 * p;          // 0..127
        gx[p]  = Xb + (size_t)(m0 + row) * K_TOT + scol;
        gw[p]  = Wt + (size_t)(n0 + row) * K_TOT + scol;
        lxp[p] = lx + row * LDS_STRIDE + scol;
        lwp[p] = lw + row * LDS_STRIDE + scol;
    }

    v8f acc[2][4] = {};

    auto issue_tile = [&](int buf, int k0) {
#pragma unroll
        for (int p = 0; p < 4; ++p)
            async_ldg_b128(lxp[p] + buf * bufX, gx[p] + k0);
#pragma unroll
        for (int p = 0; p < 4; ++p)
            async_ldg_b128(lwp[p] + buf * bufW, gw[p] + k0);
    };

    auto compute_tile = [&](int buf) {
        const bf16_t* lxb = lx + buf * bufX;
        const bf16_t* lwb = lw + buf * bufW;
#pragma unroll
        for (int ks = 0; ks < TILE_K; ks += 32) {
            // A fragments (16x32 bf16): runs at K=ks+8*hl and K=ks+16+8*hl
            v16bf afrag[2];
#pragma unroll
            for (int fm = 0; fm < 2; ++fm) {
                const bf16_t* ar = lxb + (wm * 32 + fm * 16 + l15) * LDS_STRIDE + ks;
                v8bf alo = *(const v8bf*)(ar + 8 * hl);
                v8bf ahi = *(const v8bf*)(ar + 16 + 8 * hl);
                afrag[fm] = cat8(alo, ahi);
            }
            // B fragments (32x16 bf16): runs at K=ks+8*hl and K=ks+16+8*hl
            v16bf bfrag[4];
#pragma unroll
            for (int fn = 0; fn < 4; ++fn) {
                const bf16_t* br = lwb + (wn * 64 + fn * 16 + l15) * LDS_STRIDE + ks;
                v8bf blo = *(const v8bf*)(br + 8 * hl);
                v8bf bhi = *(const v8bf*)(br + 16 + 8 * hl);
                bfrag[fn] = cat8(blo, bhi);
            }
#pragma unroll
            for (int fm = 0; fm < 2; ++fm)
#pragma unroll
                for (int fn = 0; fn < 4; ++fn)
                    acc[fm][fn] = __builtin_amdgcn_wmma_f32_16x16x32_bf16(
                        false, afrag[fm], false, bfrag[fn],
                        (short)0, acc[fm][fn], false, false);
        }
    };

    // ---- pipelined main loop: peel the last tile so the body is branch-free ----
    issue_tile(0, 0);
    int k0 = 0;
    for (; k0 < K_TOT - TILE_K; k0 += TILE_K) {
        const int buf = (k0 / TILE_K) & 1;
        issue_tile(buf ^ 1, k0 + TILE_K);   // next tile into other buffer
        wait_asynccnt_8();                  // drain current tile's 8 loads
        __syncthreads();
        compute_tile(buf);
        __syncthreads();                    // reads done before buffer refill
    }
    {
        const int buf = (k0 / TILE_K) & 1;
        wait_asynccnt_0();
        __syncthreads();
        compute_tile(buf);
    }

    // ---- epilogue: D layout -> VGPR j: M = j + 8*hl, N = l15 (per fragment) ----
#pragma unroll
    for (int fm = 0; fm < 2; ++fm) {
        int m = m0 + wm * 32 + fm * 16 + hl * 8;
#pragma unroll
        for (int fn = 0; fn < 4; ++fn) {
            int n = n0 + wn * 64 + fn * 16 + l15;
            float bv = bias[n];
            float* op = out + ((size_t)t * M_TOT + m) * N_TOT + n;
#pragma unroll
            for (int j = 0; j < 8; ++j)
                op[(size_t)j * N_TOT] = acc[fm][fn][j] + bv;
        }
    }
}

// ---------------------------------------------------------------------------
extern "C" void kernel_launch(void* const* d_in, const int* in_sizes, int n_in,
                              void* d_out, int out_size, void* d_ws, size_t ws_size,
                              hipStream_t stream) {
    const float* x      = (const float*)d_in[0];   // [B,S,DIN]
    const float* W      = (const float*)d_in[1];   // [DOUT,DIN]
    const float* bias   = (const float*)d_in[2];   // [DOUT]
    const float* lora_A = (const float*)d_in[3];   // [NL,R,DIN]
    const float* lora_B = (const float*)d_in[4];   // [NL,DOUT,R]
    const int*   tuner  = (const int*)d_in[5];     // [T]
    float* out = (float*)d_out;

    bf16_t* Xb   = (bf16_t*)d_ws;                                          // 32 MB
    bf16_t* Weff = (bf16_t*)((char*)d_ws + (size_t)M_TOT * K_TOT * 2);     //  8 MB

    // x -> bf16 (16M elems, 4/thread)
    cvt_x_bf16<<<(M_TOT * K_TOT) / 4 / 256, 256, 0, stream>>>(x, Xb);
    // fused weights (T*N*K = 4M elems)
    build_weff<<<(T_CNT * N_TOT * K_TOT) / 256, 256, 0, stream>>>(W, lora_A, lora_B,
                                                                  tuner, Weff);
    // batched GEMM + bias
    dim3 grid(M_TOT / TILE_M, N_TOT / TILE_N, T_CNT);
    lora_gemm<<<grid, 256, 0, stream>>>(Xb, Weff, bias, out);
}
